// MultiHeadAttention_1537598292446
// MI455X (gfx1250) — matrix operand order repaired and checked by reading, not compiled, not run on hardware
//
#include <hip/hip_runtime.h>
#include <hip/hip_bf16.h>

// ---------------------------------------------------------------------------
// MHA for MI455X (gfx1250): bf16 WMMA + Tensor Data Mover double-buffering.
// B=4, T=2048, DIM=1024, H=16, DH=64
// ---------------------------------------------------------------------------

typedef __attribute__((ext_vector_type(16))) __bf16       bf16x16;
typedef __attribute__((ext_vector_type(8)))  float        f32x8;
typedef __attribute__((ext_vector_type(4)))  unsigned int u32x4;
typedef __attribute__((ext_vector_type(8)))  int          i32x8;
typedef __attribute__((ext_vector_type(4)))  int          i32x4;

#define Bv   4
#define Tv   2048
#define DIMv 1024
#define Hv   16
#define DHv  64
#define Mrows (Bv * Tv)          // 8192
#define N_QKV (3 * Hv * DHv)     // 3072

#if defined(__gfx1250__) && __has_builtin(__builtin_amdgcn_tensor_load_to_lds) && \
    __has_builtin(__builtin_amdgcn_s_wait_tensorcnt)
#define USE_TDM 1
#else
#define USE_TDM 0
#endif

// --------------------------- WMMA fragment helpers -------------------------
union FragU { bf16x16 v; uint4 q[2]; };

// A-operand (16x32 row-major source): ISA 7.12.2 16-bit A layout.
// lane r=lane%16, hi=lane/16: elems 0..7 = row[hi*8+0..7], 8..15 = row[16+hi*8+0..7]
__device__ __forceinline__ bf16x16 load_afrag(const __bf16* rowp, int hi) {
  FragU u;
  u.q[0] = *(const uint4*)(rowp + hi * 8);
  u.q[1] = *(const uint4*)(rowp + 16 + hi * 8);
  return u.v;
}

// B-operand from B^T rows, same K order as A: elems 0..7 = row[hi*8+0..7],
// 8..15 = row[16+hi*8+0..7].
__device__ __forceinline__ bf16x16 load_bfrag(const __bf16* rowp, int hi) {
  FragU u;
  u.q[0] = *(const uint4*)(rowp + hi * 8);
  u.q[1] = *(const uint4*)(rowp + 16 + hi * 8);
  return u.v;
}

__device__ __forceinline__ f32x8 wmma_bf16(bf16x16 a, bf16x16 b, f32x8 c) {
  return __builtin_amdgcn_wmma_f32_16x16x32_bf16(
      false, a, false, b, (short)0, c, false, false);
}

// --------------------------- TDM helpers -----------------------------------
#if USE_TDM
// Low 32 bits of a generic pointer to LDS = LDS byte offset (aperture base
// occupies bits 63:32 only, per flat-address translation).
__device__ __forceinline__ unsigned lds_off(const void* p) {
  return (unsigned)(unsigned long long)p;
}

// 2D tile load: tile_d1 rows of tile_d0 bf16 elems, row stride `stride0`
// elems, packed contiguously (row-major [tile_d1][tile_d0]) at lds_byte.
__device__ __forceinline__ void tdm_load_2d(unsigned lds_byte, const void* gptr,
                                            unsigned tile_d0, unsigned tile_d1,
                                            unsigned stride0) {
  unsigned long long ga = (unsigned long long)gptr;
  u32x4 g0 = {
      1u,                                   // count=1, user mode
      lds_byte,                             // lds_addr
      (unsigned)(ga & 0xffffffffu),         // global_addr[31:0]
      (unsigned)((ga >> 32) & 0x01ffffffu) | (2u << 30)  // addr[56:32] | type=2
  };
  const unsigned td0 = 0x00100000u;         // tensor_dim0 (>= tile, no OOB)
  const unsigned td1 = 0x00100000u;         // tensor_dim1
  i32x8 g1 = {
      (int)(1u << 16),                                       // data_size=2B
      (int)((td0 & 0xffffu) << 16),                          // dim0[15:0]
      (int)(((td0 >> 16) & 0xffffu) | ((td1 & 0xffffu) << 16)),
      (int)(((td1 >> 16) & 0xffffu) | (tile_d0 << 16)),      // tile_dim0
      (int)(tile_d1 & 0xffffu),                              // tile_dim1 (dim2=0)
      (int)stride0,                                          // dim0_stride lo
      0, 0
  };
  i32x4 z4 = {0, 0, 0, 0};
  i32x8 z8 = {0, 0, 0, 0, 0, 0, 0, 0};
  __builtin_amdgcn_tensor_load_to_lds(g0, g1, z4, z4, z8, 0);
}

__device__ __forceinline__ void tdm_wait(bool more_in_flight) {
  if (more_in_flight) __builtin_amdgcn_s_wait_tensorcnt(1);
  else                __builtin_amdgcn_s_wait_tensorcnt(0);
}
#endif

// --------------------------- fp32 -> bf16 ----------------------------------
__global__ __launch_bounds__(256) void cvt_f32_to_bf16(const float* __restrict__ in,
                                                       __bf16* __restrict__ out, int n) {
  int i = blockIdx.x * 256 + threadIdx.x;
  if (i < n) out[i] = (__bf16)in[i];
}

// Transpose + convert: out[n][k] = (bf16) in[k][n].   in: [K][N] fp32.
__global__ __launch_bounds__(256) void transpose_to_bf16(const float* __restrict__ in,
                                                         __bf16* __restrict__ out,
                                                         int K, int N) {
  __shared__ float tile[32][33];
  int k0 = blockIdx.x * 32, n0 = blockIdx.y * 32;
  int tx = threadIdx.x & 31, ty = threadIdx.x >> 5;  // 32 x 8
#pragma unroll
  for (int i = ty; i < 32; i += 8) tile[i][tx] = in[(size_t)(k0 + i) * N + n0 + tx];
  __syncthreads();
#pragma unroll
  for (int i = ty; i < 32; i += 8)
    out[(size_t)(n0 + i) * K + k0 + tx] = (__bf16)tile[tx][i];
}

// ---------------------------------------------------------------------------
// Tiled bf16 GEMM core: C(64x128 per WG) = A[M][1024] * Bt[N][1024]^T
// 8 waves, each computes 32x32 (2x2 WMMA accums). Double-buffered LDS,
// tiles staged by TDM (waves 0/1) or cooperative b128 copies.
// ---------------------------------------------------------------------------
__device__ __forceinline__ void gemm_core(const __bf16* __restrict__ A,
                                          const __bf16* __restrict__ Bt,
                                          __bf16* Alds,  // [2][64*32]
                                          __bf16* Blds,  // [2][128*32]
                                          f32x8 (&acc)[2][2], int bm, int bn) {
  const int tid  = threadIdx.x;
  const int lane = tid & 31;
  const int wave = tid >> 5;
  const int wm = wave & 1, wn = wave >> 1;
  const int r = lane & 15, hi = lane >> 4;
  const int NSTEP = DIMv / 32;

  auto stage = [&](int s) {
    int b = s & 1;
    int k = s * 32;
#if USE_TDM
    if (wave == 0)
      tdm_load_2d(lds_off(Alds + b * (64 * 32)),
                  A + (size_t)bm * DIMv + k, 32u, 64u, DIMv);
    else if (wave == 1)
      tdm_load_2d(lds_off(Blds + b * (128 * 32)),
                  Bt + (size_t)bn * DIMv + k, 32u, 128u, DIMv);
#else
    {
      int row = tid >> 2, col = (tid & 3) * 8;
      *(uint4*)&Alds[b * (64 * 32) + row * 32 + col] =
          *(const uint4*)&A[(size_t)(bm + row) * DIMv + k + col];
    }
    {
      int row = tid >> 1, col = (tid & 1) * 16;
      const __bf16* src = &Bt[(size_t)(bn + row) * DIMv + k + col];
      *(uint4*)&Blds[b * (128 * 32) + row * 32 + col]     = *(const uint4*)(src);
      *(uint4*)&Blds[b * (128 * 32) + row * 32 + col + 8] = *(const uint4*)(src + 8);
    }
#endif
  };

  stage(0);
  for (int s = 0; s < NSTEP; ++s) {
    int cur = s & 1;
    bool more = (s + 1 < NSTEP);
    if (more) stage(s + 1);
#if USE_TDM
    tdm_wait(more);
#endif
    __syncthreads();

    const __bf16* Ab = Alds + cur * (64 * 32);
    const __bf16* Bb = Blds + cur * (128 * 32);
    bf16x16 af[2], bfr[2];
#pragma unroll
    for (int mt = 0; mt < 2; ++mt)
      af[mt] = load_afrag(&Ab[(wm * 32 + mt * 16 + r) * 32], hi);
#pragma unroll
    for (int nt = 0; nt < 2; ++nt)
      bfr[nt] = load_bfrag(&Bb[(wn * 32 + nt * 16 + r) * 32], hi);
#pragma unroll
    for (int mt = 0; mt < 2; ++mt)
#pragma unroll
      for (int nt = 0; nt < 2; ++nt)
        acc[mt][nt] = wmma_bf16(af[mt], bfr[nt], acc[mt][nt]);
    __syncthreads();
  }
}

// ---------------------------------------------------------------------------
// QKV projection. Column c of 3072 decomposes as c = d*48 + which*16 + h.
// Q,K stored [b,h,t,dh]; V stored TRANSPOSED [b,h,dh,t] for the PV WMMA.
// ---------------------------------------------------------------------------
__global__ __launch_bounds__(256) void gemm_qkv(const __bf16* __restrict__ A,
                                                const __bf16* __restrict__ Bt,
                                                __bf16* __restrict__ Qo,
                                                __bf16* __restrict__ Ko,
                                                __bf16* __restrict__ Vto) {
  __shared__ __bf16 Alds[2][64 * 32];
  __shared__ __bf16 Blds[2][128 * 32];
  const int bm = blockIdx.x * 64;
  const int bn = blockIdx.y * 128;
  f32x8 acc[2][2] = {};
  gemm_core(A, Bt, &Alds[0][0], &Blds[0][0], acc, bm, bn);

  const int lane = threadIdx.x & 31;
  const int wave = threadIdx.x >> 5;
  const int wm = wave & 1, wn = wave >> 1;
  const int r = lane & 15, hi = lane >> 4;
#pragma unroll
  for (int mt = 0; mt < 2; ++mt)
#pragma unroll
    for (int nt = 0; nt < 2; ++nt)
#pragma unroll
      for (int v = 0; v < 8; ++v) {
        int row = bm + wm * 32 + mt * 16 + v + 8 * hi;
        int col = bn + wn * 32 + nt * 16 + r;
        int d = col / 48, rem = col % 48;
        int which = rem >> 4, h = rem & 15;
        int b = row >> 11, t = row & (Tv - 1);
        int bh = b * Hv + h;
        __bf16 val = (__bf16)acc[mt][nt][v];
        if (which == 0)      Qo[((size_t)bh * Tv + t) * DHv + d] = val;
        else if (which == 1) Ko[((size_t)bh * Tv + t) * DHv + d] = val;
        else                 Vto[((size_t)bh * DHv + d) * Tv + t] = val;
      }
}

// ---------------------------------------------------------------------------
// Output projection: out = attn @ W_o (Bt = W_o^T), fp32 epilogue.
// ---------------------------------------------------------------------------
__global__ __launch_bounds__(256) void gemm_out(const __bf16* __restrict__ A,
                                                const __bf16* __restrict__ Bt,
                                                float* __restrict__ out) {
  __shared__ __bf16 Alds[2][64 * 32];
  __shared__ __bf16 Blds[2][128 * 32];
  const int bm = blockIdx.x * 64;
  const int bn = blockIdx.y * 128;
  f32x8 acc[2][2] = {};
  gemm_core(A, Bt, &Alds[0][0], &Blds[0][0], acc, bm, bn);

  const int lane = threadIdx.x & 31;
  const int wave = threadIdx.x >> 5;
  const int wm = wave & 1, wn = wave >> 1;
  const int r = lane & 15, hi = lane >> 4;
#pragma unroll
  for (int mt = 0; mt < 2; ++mt)
#pragma unroll
    for (int nt = 0; nt < 2; ++nt)
#pragma unroll
      for (int v = 0; v < 8; ++v) {
        int row = bm + wm * 32 + mt * 16 + v + 8 * hi;
        int col = bn + wn * 32 + nt * 16 + r;
        out[(size_t)row * DIMv + col] = acc[mt][nt][v];
      }
}

// ---------------------------------------------------------------------------
// Flash attention. 4 waves/WG share one (b,h); each wave owns 16 queries.
// K block [32][64] and Vt block [64][32] staged in double-buffered LDS via
// TDM (waves 0/1). Per 32-key block: 4 WMMAs for S, online softmax with
// 16-lane shfl_xor reductions, P->LDS, 4 WMMAs for O += P*V.
// ---------------------------------------------------------------------------
__global__ __launch_bounds__(128) void attn_kernel(const __bf16* __restrict__ Q,
                                                   const __bf16* __restrict__ K,
                                                   const __bf16* __restrict__ Vt,
                                                   __bf16* __restrict__ Aout) {
  __shared__ __bf16 Klds[2][32 * 64];    // [j][d]
  __shared__ __bf16 Vtlds[2][64 * 32];   // [d][j]
  __shared__ __bf16 Plds[4][16 * 32];    // per-wave P tile [m][j]

  const int tid = threadIdx.x, lane = tid & 31, wave = tid >> 5;
  const int bh = blockIdx.x >> 5;
  const int qgroup = blockIdx.x & 31;
  const int b = bh >> 4, h = bh & 15;
  const int q0 = (qgroup * 4 + wave) * 16;
  const int r = lane & 15, hi = lane >> 4;

  const __bf16* Qb  = Q  + (size_t)bh * Tv * DHv;
  const __bf16* Kb  = K  + (size_t)bh * Tv * DHv;
  const __bf16* Vtb = Vt + (size_t)bh * DHv * Tv;   // [64][2048]

  // Persistent Q fragments (A operand): rows q0+r, k-chunks 0 and 32.
  bf16x16 qf0 = load_afrag(Qb + (size_t)(q0 + r) * DHv + 0, hi);
  bf16x16 qf1 = load_afrag(Qb + (size_t)(q0 + r) * DHv + 32, hi);

  f32x8 oacc[4] = {};
  float mrun[8], lrun[8];
#pragma unroll
  for (int v = 0; v < 8; ++v) { mrun[v] = -1e30f; lrun[v] = 0.0f; }
  const float scale = 0.125f;  // DH^-0.5
  const int NBLK = Tv / 32;

  auto stage = [&](int s) {
    int bb = s & 1;
    int j = s * 32;
#if USE_TDM
    if (wave == 0)
      tdm_load_2d(lds_off(&Klds[bb][0]), Kb + (size_t)j * DHv, 64u, 32u, DHv);
    else if (wave == 1)
      tdm_load_2d(lds_off(&Vtlds[bb][0]), Vtb + j, 32u, 64u, Tv);
#else
    {
      int row = tid >> 2, col = (tid & 3) * 16;   // K: 32 rows x 64
      const __bf16* src = &Kb[(size_t)(j + row) * DHv + col];
      *(uint4*)&Klds[bb][row * 64 + col]     = *(const uint4*)(src);
      *(uint4*)&Klds[bb][row * 64 + col + 8] = *(const uint4*)(src + 8);
    }
    {
      int row = tid >> 1, col = (tid & 1) * 16;   // Vt: 64 rows x 32
      const __bf16* src = &Vtb[(size_t)row * Tv + j + col];
      *(uint4*)&Vtlds[bb][row * 32 + col]     = *(const uint4*)(src);
      *(uint4*)&Vtlds[bb][row * 32 + col + 8] = *(const uint4*)(src + 8);
    }
#endif
  };

  stage(0);
  for (int s = 0; s < NBLK; ++s) {
    int cur = s & 1;
    bool more = (s + 1 < NBLK);
    if (more) stage(s + 1);
#if USE_TDM
    tdm_wait(more);
#endif
    __syncthreads();

    // S = Q * K^T : two 16x16 tiles covering 32 keys; K-dim = DH = 2x32.
    f32x8 sv[2];
#pragma unroll
    for (int jh = 0; jh < 2; ++jh) {
      f32x8 z = {};
      bf16x16 b0 = load_bfrag(&Klds[cur][(jh * 16 + r) * 64 + 0], hi);
      bf16x16 b1 = load_bfrag(&Klds[cur][(jh * 16 + r) * 64 + 32], hi);
      z = wmma_bf16(qf0, b0, z);
      z = wmma_bf16(qf1, b1, z);
      sv[jh] = z;
    }

    // Online softmax (row = 16 lanes of one half-wave in the C/D layout).
    float alpha[8];
#pragma unroll
    for (int v = 0; v < 8; ++v) {
      float s0 = sv[0][v] * scale, s1 = sv[1][v] * scale;
      float tm = fmaxf(s0, s1);
      tm = fmaxf(tm, __shfl_xor(tm, 1));
      tm = fmaxf(tm, __shfl_xor(tm, 2));
      tm = fmaxf(tm, __shfl_xor(tm, 4));
      tm = fmaxf(tm, __shfl_xor(tm, 8));
      float mnew = fmaxf(mrun[v], tm);
      alpha[v] = __expf(mrun[v] - mnew);
      float p0 = __expf(s0 - mnew), p1 = __expf(s1 - mnew);
      sv[0][v] = p0; sv[1][v] = p1;
      float rs = p0 + p1;
      rs += __shfl_xor(rs, 1);
      rs += __shfl_xor(rs, 2);
      rs += __shfl_xor(rs, 4);
      rs += __shfl_xor(rs, 8);
      lrun[v] = lrun[v] * alpha[v] + rs;
      mrun[v] = mnew;
    }

    // Rescale O, spill P (bf16) into per-wave A-layout tile [16][32].
    __bf16* pb = &Plds[wave][0];
#pragma unroll
    for (int v = 0; v < 8; ++v) {
      int m = v + 8 * hi;
      pb[m * 32 + r]      = (__bf16)sv[0][v];
      pb[m * 32 + 16 + r] = (__bf16)sv[1][v];
#pragma unroll
      for (int dt = 0; dt < 4; ++dt) oacc[dt][v] *= alpha[v];
    }

    // O += P * V  (B^T rows = Vt rows, contiguous b128 loads).
    bf16x16 pf = load_afrag(&pb[r * 32], hi);
#pragma unroll
    for (int dt = 0; dt < 4; ++dt) {
      bf16x16 vf = load_bfrag(&Vtlds[cur][(dt * 16 + r) * 32], hi);
      oacc[dt] = wmma_bf16(pf, vf, oacc[dt]);
    }
    __syncthreads();
  }

  // Finalize: divide by l, store as [b*T][H*DH] ('b h t d -> b t (h d)').
#pragma unroll
  for (int v = 0; v < 8; ++v) {
    float inv = 1.0f / lrun[v];
    int m = v + 8 * hi;
    size_t row = (size_t)b * Tv + q0 + m;
#pragma unroll
    for (int dt = 0; dt < 4; ++dt)
      Aout[row * (Hv * DHv) + h * DHv + dt * 16 + r] = (__bf16)(oacc[dt][v] * inv);
  }
}

// ---------------------------------------------------------------------------
// Host-side launch
// ---------------------------------------------------------------------------
extern "C" void kernel_launch(void* const* d_in, const int* in_sizes, int n_in,
                              void* d_out, int out_size, void* d_ws, size_t ws_size,
                              hipStream_t stream) {
  const float* x     = (const float*)d_in[0];   // [4,2048,1024]
  const float* W_qkv = (const float*)d_in[1];   // [1024,3072]
  const float* W_o   = (const float*)d_in[2];   // [1024,1024]
  float* out = (float*)d_out;                   // [4,2048,1024]

  size_t off = 0;
  auto carve = [&](size_t elems) {
    void* p = (char*)d_ws + off;
    off = (off + elems * sizeof(__bf16) + 255) & ~(size_t)255;
    return (__bf16*)p;
  };
  __bf16* xb    = carve((size_t)Mrows * DIMv);        // x bf16
  __bf16* wqkvt = carve((size_t)N_QKV * DIMv);        // W_qkv^T bf16 [3072][1024]
  __bf16* wot   = carve((size_t)DIMv * DIMv);         // W_o^T bf16 [1024][1024]
  __bf16* Qb    = carve((size_t)Bv * Hv * Tv * DHv);  // Q  [b,h,t,dh]
  __bf16* Kb    = carve((size_t)Bv * Hv * Tv * DHv);  // K  [b,h,t,dh]
  __bf16* Vtb   = carve((size_t)Bv * Hv * DHv * Tv);  // Vt [b,h,dh,t]
  __bf16* aout  = carve((size_t)Mrows * DIMv);        // attn out [b*t][h*dh]

  // 1) conversions / weight transposes
  {
    int n = Mrows * DIMv;
    cvt_f32_to_bf16<<<(n + 255) / 256, 256, 0, stream>>>(x, xb, n);
  }
  transpose_to_bf16<<<dim3(DIMv / 32, N_QKV / 32), 256, 0, stream>>>(W_qkv, wqkvt, DIMv, N_QKV);
  transpose_to_bf16<<<dim3(DIMv / 32, DIMv / 32), 256, 0, stream>>>(W_o, wot, DIMv, DIMv);

  // 2) QKV projection (WMMA GEMM; V emitted transposed)
  gemm_qkv<<<dim3(Mrows / 64, N_QKV / 128), 256, 0, stream>>>(xb, wqkvt, Qb, Kb, Vtb);

  // 3) Flash attention (TDM-staged K/V, WMMA QK^T and PV)
  attn_kernel<<<(Bv * Hv) * (Tv / 64), 128, 0, stream>>>(Qb, Kb, Vtb, aout);

  // 4) Output projection
  gemm_out<<<dim3(Mrows / 64, DIMv / 128), 256, 0, stream>>>(aout, wot, out);
}
